// MoDACausalSelfAttention_4526895530165
// MI455X (gfx1250) — compile-verified, matrix operand order checked by reading, not run on hardware
//
#include <hip/hip_runtime.h>
#include <math.h>

#define Bq   2
#define Tq   2048
#define Cq   2048
#define Hq   16
#define HKq  4
#define Dq   128
#define NREPq (Hq / HKq)

typedef _Float16 half_t;
typedef __attribute__((ext_vector_type(16))) _Float16 v16h;
typedef __attribute__((ext_vector_type(8)))  _Float16 v8h;
typedef __attribute__((ext_vector_type(4)))  _Float16 v4h;
typedef __attribute__((ext_vector_type(8)))  float    v8f;
typedef __attribute__((ext_vector_type(4)))  unsigned int v4u;
typedef __attribute__((ext_vector_type(8)))  int      v8i;
typedef __attribute__((ext_vector_type(4)))  int      v4i;

#if __has_builtin(__builtin_amdgcn_tensor_load_to_lds) && __has_builtin(__builtin_amdgcn_s_wait_tensorcnt)
#define HAVE_TDM 1
#else
#define HAVE_TDM 0
#endif

// Build a 16x32 f16 A/B fragment per ISA 7.12.2: per-lane contiguous row/col,
// regs0-3 = K[sel8..sel8+7], regs4-7 = K[16+sel8..+7] (sel8 = lanes>=16 ? 8 : 0)
__device__ __forceinline__ v16h frag_ld(const half_t* p, int sel8) {
  v8h lo = *(const v8h*)(p + sel8);
  v8h hi = *(const v8h*)(p + 16 + sel8);
  return __builtin_shufflevector(lo, hi, 0,1,2,3,4,5,6,7,8,9,10,11,12,13,14,15);
}

__device__ __forceinline__ unsigned lds_off32(const void* p) {
  // flat shared address = {SHARED_BASE, lds_offset32}; low dword is the LDS byte address
  return (unsigned)(uintptr_t)p;
}

// Issue one 2D-tile TDM load: tile_x contiguous f16 elems per row, tile_y rows,
// row stride `stride` (elements). D# per CDNA5 ISA ch.8 (type=2, data_size=2B).
__device__ __forceinline__ void tdm_issue_2d(unsigned lds_addr, const void* g,
                                             unsigned tx, unsigned ty,
                                             unsigned long long stride) {
#if HAVE_TDM
  unsigned long long ga = (unsigned long long)(uintptr_t)g;
  v4u g0;
  g0.x = 1u;                                            // count=1, user mode
  g0.y = lds_addr;                                      // lds_addr [63:32]
  g0.z = (unsigned)ga;                                  // global_addr [95:64]
  g0.w = (unsigned)((ga >> 32) & 0x01FFFFFFu) | (2u << 30);  // addr[56:32], type=2
  v8i g1;
  g1[0] = (int)(1u << 16);                              // data_size = 2 bytes
  g1[1] = (int)((tx & 0xFFFFu) << 16);                  // tensor_dim0[15:0]
  g1[2] = (int)(((tx >> 16) & 0xFFFFu) | ((ty & 0xFFFFu) << 16));   // td0 hi | td1 lo
  g1[3] = (int)(((ty >> 16) & 0xFFFFu) | ((tx & 0xFFFFu) << 16));   // td1 hi | tile_dim0
  g1[4] = (int)(ty & 0xFFFFu);                          // tile_dim1 (tile_dim2 = 0)
  g1[5] = (int)(unsigned)(stride & 0xFFFFFFFFull);      // tensor_dim0_stride lo32
  g1[6] = (int)(unsigned)((stride >> 32) & 0xFFFFull);  // tensor_dim0_stride hi16
  g1[7] = 0;
  v4i z4 = {};
#if __clang_major__ >= 23
  v8i z8 = {};
  __builtin_amdgcn_tensor_load_to_lds(g0, g1, z4, z4, z8, 0);
#else
  __builtin_amdgcn_tensor_load_to_lds(g0, g1, z4, z4, 0);
#endif
#endif
}

// ---------------- f32 -> f16 conversion ----------------
__global__ void cvt_f32_f16(const float* __restrict__ in, half_t* __restrict__ out, int n4) {
  int i = blockIdx.x * blockDim.x + threadIdx.x;
  int stride = gridDim.x * blockDim.x;
  for (; i < n4; i += stride) {
    float4 v = ((const float4*)in)[i];
    v4h h;
    h.x = (half_t)v.x; h.y = (half_t)v.y; h.z = (half_t)v.z; h.w = (half_t)v.w;
    ((v4h*)out)[i] = h;
  }
}

// ---------------- tiled f32 (K,N) -> f16 (N,K) transpose ----------------
__global__ void cvt_transpose_f32_f16(const float* __restrict__ in, half_t* __restrict__ out,
                                      int K, int N) {
  __shared__ float tile[32][33];
  int n0 = blockIdx.x * 32, k0 = blockIdx.y * 32;
  int c = threadIdx.x, r = threadIdx.y;          // blockDim (32, 8)
  for (int rr = r; rr < 32; rr += 8)
    tile[rr][c] = in[(size_t)(k0 + rr) * N + n0 + c];
  __syncthreads();
  for (int rr = r; rr < 32; rr += 8)
    out[(size_t)(n0 + rr) * K + k0 + c] = (half_t)tile[c][rr];
}

// ---------------- WMMA GEMM, TDM double-buffered LDS tiles, fused RoPE epilogue ----
// C(MxN) = A(MxK,f16 row-major) * Bt^T, Bt pre-transposed: (N,K) f16.
// mode 0: row-major store (oh0 f16 if non-null, else of0 f32); rope==1 -> RoPE all cols
// mode 3: fused k|v|dk|dv routing epilogue, N = 4*512:
//   which = gc>>9:  0 -> oh0 (B,HK,T,D) f16 with RoPE
//                   1 -> oh1 (B,HK,D,T) f16 (V^T)
//                   2 -> of1 (B,HK,T,D) f32    3 -> of2 (B,HK,T,D) f32
__global__ void __launch_bounds__(256)
gemm_wmma_f16(const half_t* __restrict__ A, const half_t* __restrict__ Bt,
              float* __restrict__ of0, half_t* __restrict__ oh0,
              half_t* __restrict__ oh1, float* __restrict__ of1, float* __restrict__ of2,
              const float* __restrict__ fcos, const float* __restrict__ fsin,
              int M, int N, int K, int mode, int rope) {
  __shared__ __align__(16) half_t As[2][128 * 32];
  __shared__ __align__(16) half_t Bs[2][128 * 32];
  const int tid = threadIdx.x;
  const int m0 = blockIdx.y * 128, n0 = blockIdx.x * 128;
  const int wid = tid >> 5, ln = tid & 31;
  const int lr = ln & 15;
  const int sel8 = (ln & 16) ? 8 : 0;
  const int mbase = (wid & 3) * 32, nbase = (wid >> 2) * 64;

  v8f acc[2][4];
  v8f zacc = {};
  for (int a = 0; a < 2; a++)
    for (int bb = 0; bb < 4; bb++) acc[a][bb] = zacc;

  // prologue: stage first K-slice into buffer 0
#if HAVE_TDM
  if (wid == 0) {
    tdm_issue_2d(lds_off32(&As[0][0]), &A[(size_t)m0 * K], 32, 128, (unsigned long long)K);
    tdm_issue_2d(lds_off32(&Bs[0][0]), &Bt[(size_t)n0 * K], 32, 128, (unsigned long long)K);
  }
#else
  for (int c = tid; c < 512; c += 256) {
    int row = c >> 2, col = (c & 3) << 3;
    *(v8h*)&As[0][row * 32 + col] = *(const v8h*)&A[(size_t)(m0 + row) * K + col];
    *(v8h*)&Bs[0][row * 32 + col] = *(const v8h*)&Bt[(size_t)(n0 + row) * K + col];
  }
#endif

  int cur = 0;
  for (int kk = 0; kk < K; kk += 32) {
    const bool more = (kk + 32) < K;
#if HAVE_TDM
    if (wid == 0) {
      if (more) {
        tdm_issue_2d(lds_off32(&As[cur ^ 1][0]), &A[(size_t)m0 * K + kk + 32], 32, 128,
                     (unsigned long long)K);
        tdm_issue_2d(lds_off32(&Bs[cur ^ 1][0]), &Bt[(size_t)n0 * K + kk + 32], 32, 128,
                     (unsigned long long)K);
        __builtin_amdgcn_s_wait_tensorcnt(2);   // oldest pair (cur) complete
      } else {
        __builtin_amdgcn_s_wait_tensorcnt(0);
      }
    }
#else
    if (more) {
      for (int c = tid; c < 512; c += 256) {
        int row = c >> 2, col = (c & 3) << 3;
        *(v8h*)&As[cur ^ 1][row * 32 + col] =
            *(const v8h*)&A[(size_t)(m0 + row) * K + kk + 32 + col];
        *(v8h*)&Bs[cur ^ 1][row * 32 + col] =
            *(const v8h*)&Bt[(size_t)(n0 + row) * K + kk + 32 + col];
      }
    }
#endif
    __syncthreads();

    v16h af0 = frag_ld(&As[cur][(mbase + lr) * 32], sel8);
    v16h af1 = frag_ld(&As[cur][(mbase + 16 + lr) * 32], sel8);
#pragma unroll
    for (int nt = 0; nt < 4; nt++) {
      v16h bf = frag_ld(&Bs[cur][(nbase + nt * 16 + lr) * 32], sel8);
      acc[0][nt] = __builtin_amdgcn_wmma_f32_16x16x32_f16(false, af0, false, bf,
                                                          (short)0, acc[0][nt], false, false);
      acc[1][nt] = __builtin_amdgcn_wmma_f32_16x16x32_f16(false, af1, false, bf,
                                                          (short)0, acc[1][nt], false, false);
    }
    __syncthreads();   // all waves done reading `cur` before it is refilled
    cur ^= 1;
  }

  const int mrow_off = (ln & 16) ? 8 : 0;
  const bool rope_any = (rope != 0) || (mode == 3);
#pragma unroll
  for (int mt = 0; mt < 2; mt++)
#pragma unroll
    for (int nt = 0; nt < 4; nt++)
#pragma unroll
      for (int g = 0; g < 8; g++) {
        int gr = m0 + mbase + mt * 16 + g + mrow_off;
        int gc = n0 + nbase + nt * 16 + lr;
        float val = acc[mt][nt][g];
        if (rope_any) {
          // RoPE pair (d, d^1) lives in the adjacent lane of the 16-lane group
          float partner = __shfl_xor(val, 1, 32);
          bool do_rope = (rope != 0) || ((gc >> 9) == 0);
          if (do_rope) {
            int t = gr & (Tq - 1);
            int d = gc & (Dq - 1);
            int d2 = d >> 1;
            float c = fcos[t * 64 + d2], s = fsin[t * 64 + d2];
            val = (d & 1) ? (partner * s + val * c)   // oo = xe*s + xo*c
                          : (val * c - partner * s);  // oe = xe*c - xo*s
          }
        }
        if (mode == 0) {
          size_t oidx = (size_t)gr * N + gc;
          if (oh0) oh0[oidx] = (half_t)val;
          else     of0[oidx] = val;
        } else {  // mode 3: fused k|v|dk|dv routing
          int b = gr >> 11, t = gr & (Tq - 1);
          int which = gc >> 9;
          int nn = gc & 511;
          int hk = nn >> 7, d = nn & (Dq - 1);
          size_t idx_td = (((size_t)(b * HKq + hk)) * Tq + t) * Dq + d;  // (B,HK,T,D)
          size_t idx_dt = (((size_t)(b * HKq + hk)) * Dq + d) * Tq + t;  // (B,HK,D,T)
          if (which == 0)      oh0[idx_td] = (half_t)val;  // k (roped)
          else if (which == 1) oh1[idx_dt] = (half_t)val;  // v^T
          else if (which == 2) of1[idx_td] = val;          // k_write
          else                 of2[idx_td] = val;          // v_write
        }
      }
}

// ---------------- flash attention (causal, GQA), TDM-staged K / V^T ----------------
// Q: (B,T,H,D) f16 post-RoPE, K: (B,HK,T,D) f16, VT: (B,HK,D,T) f16, Y: (B,T,H,D) f16
__global__ void __launch_bounds__(128)
attn_kernel(const half_t* __restrict__ Qm, const half_t* __restrict__ Km,
            const half_t* __restrict__ VTm, half_t* __restrict__ Ym) {
  __shared__ __align__(16) half_t Ks[2][64 * Dq];   // [key][d]  16KB each
  __shared__ __align__(16) half_t Vt[2][Dq * 64];   // [d][key]  16KB each
  __shared__ __align__(16) half_t Pl[4 * 16 * 64];  // per-wave P tile

  const int tid = threadIdx.x;
  const int wid = tid >> 5, ln = tid & 31;
  const int lr = ln & 15;
  const int hi = (ln & 16) ? 1 : 0;
  const int sel8 = hi ? 8 : 0;
  const int i0 = blockIdx.x * 64;
  const int bh = blockIdx.y;
  const int b = bh / Hq, h = bh % Hq, hk = h / NREPq;

  const size_t kvbase = ((size_t)(b * HKq + hk)) * Tq;       // K rows base
  const size_t vtbase = ((size_t)(b * HKq + hk)) * Dq * Tq;  // V^T base

  // hoist Q fragments (16 rows per wave, D=128 -> 4 K-steps)
  const half_t* qrow = Qm + (((size_t)(b * Tq + (i0 + wid * 16 + lr))) * Hq + h) * Dq;
  v16h qf[4];
#pragma unroll
  for (int ks = 0; ks < 4; ks++) qf[ks] = frag_ld(qrow + 32 * ks, sel8);

  v8f o[8];
  v8f zacc = {};
#pragma unroll
  for (int nt = 0; nt < 8; nt++) o[nt] = zacc;
  float mi[8], li[8];
#pragma unroll
  for (int g = 0; g < 8; g++) { mi[g] = -1e30f; li[g] = 0.f; }

  const float scale = 0.08838834764831845f;  // 1/sqrt(128)
  half_t* Pw = Pl + wid * 16 * 64;

  // prologue: stage j=0 tiles into buffer 0
#if HAVE_TDM
  if (wid == 0) {
    tdm_issue_2d(lds_off32(&Ks[0][0]), Km + kvbase * Dq, Dq, 64, (unsigned long long)Dq);
    tdm_issue_2d(lds_off32(&Vt[0][0]), VTm + vtbase, 64, Dq, (unsigned long long)Tq);
  }
#else
  for (int c = tid; c < 1024; c += 128) {
    int r = c >> 3, x = (c & 7) << 3;
    *(v8h*)&Ks[0][((c >> 4)) * Dq + ((c & 15) << 3)] =
        *(const v8h*)&Km[(kvbase + (c >> 4)) * Dq + ((c & 15) << 3)];
    *(v8h*)&Vt[0][r * 64 + x] = *(const v8h*)&VTm[vtbase + (size_t)r * Tq + x];
  }
#endif

  int cur = 0;
  for (int j = 0; j <= i0; j += 64) {
    const bool more = (j + 64) <= i0;
#if HAVE_TDM
    if (wid == 0) {
      if (more) {
        tdm_issue_2d(lds_off32(&Ks[cur ^ 1][0]), Km + (kvbase + j + 64) * Dq, Dq, 64,
                     (unsigned long long)Dq);
        tdm_issue_2d(lds_off32(&Vt[cur ^ 1][0]), VTm + vtbase + j + 64, 64, Dq,
                     (unsigned long long)Tq);
        __builtin_amdgcn_s_wait_tensorcnt(2);
      } else {
        __builtin_amdgcn_s_wait_tensorcnt(0);
      }
    }
#else
    if (more) {
      for (int c = tid; c < 1024; c += 128) {
        int r = c >> 3, x = (c & 7) << 3;
        *(v8h*)&Ks[cur ^ 1][((c >> 4)) * Dq + ((c & 15) << 3)] =
            *(const v8h*)&Km[(kvbase + j + 64 + (c >> 4)) * Dq + ((c & 15) << 3)];
        *(v8h*)&Vt[cur ^ 1][r * 64 + x] =
            *(const v8h*)&VTm[vtbase + (size_t)r * Tq + j + 64 + x];
      }
    }
#endif
    __syncthreads();

    // S = Q * K^T  (16x64 per wave)
    v8f s[4];
#pragma unroll
    for (int nt = 0; nt < 4; nt++) {
      s[nt] = zacc;
      const half_t* kcol = &Ks[cur][(nt * 16 + lr) * Dq];
#pragma unroll
      for (int ks = 0; ks < 4; ks++) {
        v16h kf = frag_ld(kcol + 32 * ks, sel8);
        s[nt] = __builtin_amdgcn_wmma_f32_16x16x32_f16(false, qf[ks], false, kf,
                                                       (short)0, s[nt], false, false);
      }
    }
    // scale + causal mask
#pragma unroll
    for (int nt = 0; nt < 4; nt++)
#pragma unroll
      for (int g = 0; g < 8; g++) {
        int r = i0 + wid * 16 + g + hi * 8;
        int col = j + nt * 16 + lr;
        float v = s[nt][g] * scale;
        s[nt][g] = (col <= r) ? v : -1e30f;
      }
    // online softmax (C layout: each row spans 16 lanes -> butterfly)
    float newm[8], cf[8], rs[8];
#pragma unroll
    for (int g = 0; g < 8; g++) {
      float t = s[0][g];
      t = fmaxf(t, s[1][g]); t = fmaxf(t, s[2][g]); t = fmaxf(t, s[3][g]);
#pragma unroll
      for (int msk = 1; msk < 16; msk <<= 1) t = fmaxf(t, __shfl_xor(t, msk, 32));
      newm[g] = fmaxf(mi[g], t);
      cf[g] = __expf(mi[g] - newm[g]);
      rs[g] = 0.f;
    }
#pragma unroll
    for (int nt = 0; nt < 4; nt++)
#pragma unroll
      for (int g = 0; g < 8; g++) {
        float p = __expf(s[nt][g] - newm[g]);
        rs[g] += p;
        Pw[(g + hi * 8) * 64 + nt * 16 + lr] = (half_t)p;
      }
#pragma unroll
    for (int g = 0; g < 8; g++) {
      float t = rs[g];
#pragma unroll
      for (int msk = 1; msk < 16; msk <<= 1) t += __shfl_xor(t, msk, 32);
      li[g] = li[g] * cf[g] + t;
      mi[g] = newm[g];
    }
#pragma unroll
    for (int nt = 0; nt < 8; nt++)
#pragma unroll
      for (int g = 0; g < 8; g++) o[nt][g] *= cf[g];
    // O += P * V
#pragma unroll
    for (int ks = 0; ks < 2; ks++) {
      v16h pf = frag_ld(Pw + lr * 64 + 32 * ks, sel8);
#pragma unroll
      for (int nt = 0; nt < 8; nt++) {
        v16h vf = frag_ld(&Vt[cur][(nt * 16 + lr) * 64 + 32 * ks], sel8);
        o[nt] = __builtin_amdgcn_wmma_f32_16x16x32_f16(false, pf, false, vf,
                                                       (short)0, o[nt], false, false);
      }
    }
    __syncthreads();
    cur ^= 1;
  }

  // normalize + store Y (B,T,H,D) f16
#pragma unroll
  for (int nt = 0; nt < 8; nt++)
#pragma unroll
    for (int g = 0; g < 8; g++) {
      int t = i0 + wid * 16 + g + hi * 8;
      Ym[(((size_t)(b * Tq + t)) * Hq + h) * Dq + nt * 16 + lr] = (half_t)(o[nt][g] / li[g]);
    }
}

// ---------------- host launch ----------------
extern "C" void kernel_launch(void* const* d_in, const int* in_sizes, int n_in,
                              void* d_out, int out_size, void* d_ws, size_t ws_size,
                              hipStream_t stream) {
  (void)in_sizes; (void)n_in; (void)out_size; (void)ws_size;
  const float* x      = (const float*)d_in[0];
  const float* w_q    = (const float*)d_in[1];
  const float* w_k    = (const float*)d_in[2];
  const float* w_v    = (const float*)d_in[3];
  const float* w_proj = (const float*)d_in[4];
  const float* w_dk   = (const float*)d_in[5];
  const float* w_dv   = (const float*)d_in[6];
  const float* fcos   = (const float*)d_in[7];
  const float* fsin   = (const float*)d_in[8];
  float* out = (float*)d_out;

  const size_t NX  = (size_t)Bq * Tq * Cq;        // 8388608
  const size_t NWQ = (size_t)Cq * Hq * Dq;        // 4194304
  const size_t NWK = (size_t)Cq * HKq * Dq;       // 1048576 (2MB f16: multiple of 256B)
  const size_t NQ  = (size_t)Bq * Tq * Hq * Dq;   // 8388608
  const size_t NKV = (size_t)Bq * Tq * HKq * Dq;  // 2097152
  const size_t OFF_K = NX;
  const size_t OFF_V = NX + NKV;

  char* ws = (char*)d_ws;
  size_t off = 0;
  auto carve = [&](size_t nhalf) -> half_t* {
    half_t* p = (half_t*)(ws + off);
    off += ((nhalf * sizeof(half_t) + 255) / 256) * 256;
    return p;
  };
  half_t* x_h   = carve(NX);
  half_t* wq_t  = carve(NWQ);   // transposed (N,K)
  // NOTE: the next four must stay adjacent: they form one (2048,K) concat B matrix
  half_t* wk_t  = carve(NWK);
  half_t* wv_t  = carve(NWK);
  half_t* wdk_t = carve(NWK);
  half_t* wdv_t = carve(NWK);
  half_t* wp_t  = carve(NWQ);
  half_t* q_h   = carve(NQ);
  half_t* kh_h  = carve(NKV);   // (B,HK,T,D)
  half_t* vt_h  = carve(NKV);   // (B,HK,D,T) = V^T
  half_t* y_h   = carve(NQ);

  // x: plain convert (row-major A operand)
  {
    int n4 = (int)(NX / 4);
    int blocks = (n4 + 255) / 256;
    if (blocks > 8192) blocks = 8192;
    hipLaunchKernelGGL(cvt_f32_f16, dim3(blocks), dim3(256), 0, stream, x, x_h, n4);
  }
  // weights: convert + transpose to (N,K) so GEMM B-tiles are contiguous 2D TDM tiles
  auto cvtT = [&](const float* src, half_t* dst, int K, int N) {
    hipLaunchKernelGGL(cvt_transpose_f32_f16, dim3(N / 32, K / 32), dim3(32, 8), 0, stream,
                       src, dst, K, N);
  };
  cvtT(w_q,    wq_t,  Cq, Hq * Dq);
  cvtT(w_k,    wk_t,  Cq, HKq * Dq);
  cvtT(w_v,    wv_t,  Cq, HKq * Dq);
  cvtT(w_dk,   wdk_t, Cq, HKq * Dq);
  cvtT(w_dv,   wdv_t, Cq, HKq * Dq);
  cvtT(w_proj, wp_t,  Cq, Cq);

  const int M = Bq * Tq;  // 4096
  dim3 blk(256);
  dim3 grid_full(Cq / 128, M / 128);  // (16, 32)

  // q = RoPE(x @ w_q) -> (B,T,H*D) f16  (RoPE fused in epilogue)
  hipLaunchKernelGGL(gemm_wmma_f16, grid_full, blk, 0, stream,
                     x_h, wq_t, (float*)nullptr, q_h, (half_t*)nullptr,
                     (float*)nullptr, (float*)nullptr, fcos, fsin,
                     M, Hq * Dq, Cq, 0, 1);
  // fused k|v|dk|dv over concatenated (2048,K) weights:
  //   k -> kh_h (B,HK,T,D) f16 w/ RoPE; v -> vt_h (B,HK,D,T) f16;
  //   dk -> out+OFF_K f32; dv -> out+OFF_V f32
  hipLaunchKernelGGL(gemm_wmma_f16, grid_full, blk, 0, stream,
                     x_h, wk_t, (float*)nullptr, kh_h, vt_h,
                     out + OFF_K, out + OFF_V, fcos, fsin,
                     M, 4 * HKq * Dq, Cq, 3, 0);

  // attention -> y_h (B,T,H,D) f16
  hipLaunchKernelGGL(attn_kernel, dim3(Tq / 64, Bq * Hq), dim3(128), 0, stream,
                     q_h, kh_h, vt_h, y_h);

  // out = y @ w_proj -> (B,T,C) f32 (output 1)
  hipLaunchKernelGGL(gemm_wmma_f16, grid_full, blk, 0, stream,
                     y_h, wp_t, out, (half_t*)nullptr, (half_t*)nullptr,
                     (float*)nullptr, (float*)nullptr, (const float*)nullptr,
                     (const float*)nullptr, M, Cq, Cq, 0, 0);
}
